// LightGCNWithUserAndItemInfo_1760936592044
// MI455X (gfx1250) — matrix operand order repaired and checked by reading, not compiled, not run on hardware
//
#include <hip/hip_runtime.h>
#include <cstddef>

typedef __attribute__((ext_vector_type(4)))  float        f32x4;
typedef __attribute__((ext_vector_type(8)))  float        f32x8;
typedef __attribute__((ext_vector_type(4)))  unsigned int u32x4;
typedef __attribute__((ext_vector_type(8)))  unsigned int u32x8;
typedef __attribute__((ext_vector_type(16))) __bf16       bf16x16;

__device__ __forceinline__ unsigned int pack_bf16(float lo, float hi) {
  unsigned int ul = __float_as_uint(lo);
  unsigned int uh = __float_as_uint(hi);
  ul = (ul + 0x7FFFu + ((ul >> 16) & 1u)) >> 16;   // RNE truncate to bf16
  uh = (uh + 0x7FFFu + ((uh >> 16) & 1u)) >> 16;
  return (ul & 0xFFFFu) | (uh << 16);
}

// ---------------------------------------------------------------------------
// Split-K partial GEMM: Cpart[y][M,64] = A[M, kb:ke] @ B[kb:ke, 64]
// A = adj (row-major, NT streamed). blockIdx.y = K-split id.
// WG: 256 thr = 8 waves; tile 128(M) x 64(N) x 32(K-step), K tail peeled.
// ---------------------------------------------------------------------------
__global__ __launch_bounds__(256) void gcn_gemm_ab(
    const float* __restrict__ A, const float* __restrict__ Bm,
    float* __restrict__ Cpart, int M, int K, int kspan) {
  __shared__ unsigned int lds_a[8][32][8] __attribute__((aligned(32))); // 8KB
  __shared__ unsigned int lds_b[4][32][8] __attribute__((aligned(32))); // 4KB

  const int tid  = threadIdx.x;
  const int m0   = blockIdx.x * 128;
  const int lane = tid & 31;
  const int w    = tid >> 5;

  const int kb = blockIdx.y * kspan;
  int ke = kb + kspan; if (ke > K) ke = K;

  f32x8 acc[4] = {};

  // A-staging: thread -> (row, half of 32 K)
  const int ar = tid >> 1, ah = tid & 1;
  int arow = m0 + ar; if (arow >= M) arow = M - 1;
  const float* aptr = A + (size_t)arow * K;
  const int am = ar & 15, ablk = ar >> 4;

  // B-staging: thread -> (K pair, 4 N cols)
  const int bkp = tid >> 4;
  const int bnb = (tid & 15) * 4;
  const int bnt = bnb >> 4;
  const int blb = (bnb & 15) + (((bkp >> 2) & 1) ? 16 : 0);
  const int bvg = (bkp & 3) + ((bkp >= 8) ? 4 : 0);

  auto compute = [&]() {
    u32x8 afr = *(const u32x8*)&lds_a[w][lane][0];
    u32x8 b0  = *(const u32x8*)&lds_b[0][lane][0];
    u32x8 b1  = *(const u32x8*)&lds_b[1][lane][0];
    u32x8 b2  = *(const u32x8*)&lds_b[2][lane][0];
    u32x8 b3  = *(const u32x8*)&lds_b[3][lane][0];
    bf16x16 av = __builtin_bit_cast(bf16x16, afr);
    acc[0] = __builtin_amdgcn_wmma_f32_16x16x32_bf16(
        false, av, false, __builtin_bit_cast(bf16x16, b0), (short)0, acc[0], false, false);
    acc[1] = __builtin_amdgcn_wmma_f32_16x16x32_bf16(
        false, av, false, __builtin_bit_cast(bf16x16, b1), (short)0, acc[1], false, false);
    acc[2] = __builtin_amdgcn_wmma_f32_16x16x32_bf16(
        false, av, false, __builtin_bit_cast(bf16x16, b2), (short)0, acc[2], false, false);
    acc[3] = __builtin_amdgcn_wmma_f32_16x16x32_bf16(
        false, av, false, __builtin_bit_cast(bf16x16, b3), (short)0, acc[3], false, false);
  };

  int k0 = kb;
  for (; k0 + 32 <= ke; k0 += 32) {   // hot loop: no guards at all
    __syncthreads();
    { // stage A: 128x32, full vector NT loads
      const f32x4* p = (const f32x4*)(aptr + k0 + ah * 16);
      f32x4 x0 = __builtin_nontemporal_load(p + 0);
      f32x4 x1 = __builtin_nontemporal_load(p + 1);
      f32x4 x2 = __builtin_nontemporal_load(p + 2);
      f32x4 x3 = __builtin_nontemporal_load(p + 3);
      u32x4 lo4 = { pack_bf16(x0.x,x0.y), pack_bf16(x0.z,x0.w),
                    pack_bf16(x1.x,x1.y), pack_bf16(x1.z,x1.w) };
      u32x4 hi4 = { pack_bf16(x2.x,x2.y), pack_bf16(x2.z,x2.w),
                    pack_bf16(x3.x,x3.y), pack_bf16(x3.z,x3.w) };
      *(u32x4*)&lds_a[ablk][am][ah * 4]      = lo4;
      *(u32x4*)&lds_a[ablk][am + 16][ah * 4] = hi4;
    }
    { // stage B: 32x64
      int k = k0 + 2 * bkp;
      f32x4 b0 = *(const f32x4*)(Bm + (size_t)k * 64 + bnb);
      f32x4 b1 = *(const f32x4*)(Bm + (size_t)(k + 1) * 64 + bnb);
      lds_b[bnt][blb + 0][bvg] = pack_bf16(b0.x, b1.x);
      lds_b[bnt][blb + 1][bvg] = pack_bf16(b0.y, b1.y);
      lds_b[bnt][blb + 2][bvg] = pack_bf16(b0.z, b1.z);
      lds_b[bnt][blb + 3][bvg] = pack_bf16(b0.w, b1.w);
    }
    __syncthreads();
    compute();
  }

  if (k0 < ke) {                      // peeled K tail (runs at most once)
    __syncthreads();
    {
      int kbase = k0 + ah * 16;
      float v[16];
      for (int t = 0; t < 16; ++t) { int k = kbase + t; v[t] = (k < ke) ? aptr[k] : 0.0f; }
      u32x4 lo4 = { pack_bf16(v[0],v[1]),  pack_bf16(v[2],v[3]),
                    pack_bf16(v[4],v[5]),  pack_bf16(v[6],v[7]) };
      u32x4 hi4 = { pack_bf16(v[8],v[9]),  pack_bf16(v[10],v[11]),
                    pack_bf16(v[12],v[13]),pack_bf16(v[14],v[15]) };
      *(u32x4*)&lds_a[ablk][am][ah * 4]      = lo4;
      *(u32x4*)&lds_a[ablk][am + 16][ah * 4] = hi4;
    }
    {
      int k = k0 + 2 * bkp;
      f32x4 b0 = {0.f,0.f,0.f,0.f}, b1 = {0.f,0.f,0.f,0.f};
      if (k < ke)     b0 = *(const f32x4*)(Bm + (size_t)k * 64 + bnb);
      if (k + 1 < ke) b1 = *(const f32x4*)(Bm + (size_t)(k + 1) * 64 + bnb);
      lds_b[bnt][blb + 0][bvg] = pack_bf16(b0.x, b1.x);
      lds_b[bnt][blb + 1][bvg] = pack_bf16(b0.y, b1.y);
      lds_b[bnt][blb + 2][bvg] = pack_bf16(b0.z, b1.z);
      lds_b[bnt][blb + 3][bvg] = pack_bf16(b0.w, b1.w);
    }
    __syncthreads();
    compute();
  }

  float* Cp = Cpart + (size_t)blockIdx.y * ((size_t)M * 64);
  const int nlo   = lane & 15;
  const int mbase = m0 + w * 16 + ((lane & 16) ? 8 : 0);
#pragma unroll
  for (int nt = 0; nt < 4; ++nt) {
    int n = nt * 16 + nlo;
#pragma unroll
    for (int r = 0; r < 8; ++r) {
      int m = mbase + r;
      if (m < M) Cp[(size_t)m * 64 + n] = acc[nt][r];
    }
  }
}

// ---------------------------------------------------------------------------
// Split-K partial: Cpart[y][item,d] = sum_{k in [kb,ke)} U[k,d]*Adj[k,item]
// (== (adj.T @ U) tile). adj is the B operand -> coalesced row streaming.
// ---------------------------------------------------------------------------
__global__ __launch_bounds__(256) void gcn_gemm_atb(
    const float* __restrict__ Adj, const float* __restrict__ U,
    float* __restrict__ Cpart, int K, int Nit, int kspan) {
  __shared__ unsigned int lds_a[4][32][8] __attribute__((aligned(32))); // 4KB
  __shared__ unsigned int lds_b[8][32][8] __attribute__((aligned(32))); // 8KB

  const int tid  = threadIdx.x;
  const int n0   = blockIdx.x * 128;
  const int lane = tid & 31;
  const int w    = tid >> 5;

  const int kb = blockIdx.y * kspan;
  int ke = kb + kspan; if (ke > K) ke = K;

  f32x8 acc[4] = {};

  const int kp  = tid >> 4;           // K pair 0..15
  const int adb = (tid & 15) * 4;     // A-stage d cols
  const int amt = adb >> 4, amb = adb & 15;
  const int alo = ((kp >> 2) & 1) ? 16 : 0;
  const int avg = (kp & 3) + ((kp >= 8) ? 4 : 0);

  const int bnb = (tid & 15) * 8;     // B-stage item cols (local)
  const int bnt = bnb >> 4;
  const int blb = (bnb & 15) + (((kp >> 2) & 1) ? 16 : 0);
  const int bvg = (kp & 3) + ((kp >= 8) ? 4 : 0);
  const bool edge = (n0 + 128 > Nit);

  auto compute = [&]() {
    u32x8 bfr = *(const u32x8*)&lds_b[w][lane][0];
    u32x8 a0  = *(const u32x8*)&lds_a[0][lane][0];
    u32x8 a1  = *(const u32x8*)&lds_a[1][lane][0];
    u32x8 a2  = *(const u32x8*)&lds_a[2][lane][0];
    u32x8 a3  = *(const u32x8*)&lds_a[3][lane][0];
    bf16x16 bv = __builtin_bit_cast(bf16x16, bfr);
    acc[0] = __builtin_amdgcn_wmma_f32_16x16x32_bf16(
        false, __builtin_bit_cast(bf16x16, a0), false, bv, (short)0, acc[0], false, false);
    acc[1] = __builtin_amdgcn_wmma_f32_16x16x32_bf16(
        false, __builtin_bit_cast(bf16x16, a1), false, bv, (short)0, acc[1], false, false);
    acc[2] = __builtin_amdgcn_wmma_f32_16x16x32_bf16(
        false, __builtin_bit_cast(bf16x16, a2), false, bv, (short)0, acc[2], false, false);
    acc[3] = __builtin_amdgcn_wmma_f32_16x16x32_bf16(
        false, __builtin_bit_cast(bf16x16, a3), false, bv, (short)0, acc[3], false, false);
  };

  auto stage_b = [&](const float* r0, const float* r1) {
    int ibase = n0 + bnb;
    float x0[8], x1[8];
    if (!edge) {
      f32x4 p0 = __builtin_nontemporal_load((const f32x4*)(r0 + ibase));
      f32x4 p1 = __builtin_nontemporal_load((const f32x4*)(r0 + ibase + 4));
      f32x4 q0 = __builtin_nontemporal_load((const f32x4*)(r1 + ibase));
      f32x4 q1 = __builtin_nontemporal_load((const f32x4*)(r1 + ibase + 4));
      x0[0]=p0.x; x0[1]=p0.y; x0[2]=p0.z; x0[3]=p0.w;
      x0[4]=p1.x; x0[5]=p1.y; x0[6]=p1.z; x0[7]=p1.w;
      x1[0]=q0.x; x1[1]=q0.y; x1[2]=q0.z; x1[3]=q0.w;
      x1[4]=q1.x; x1[5]=q1.y; x1[6]=q1.z; x1[7]=q1.w;
    } else {
      for (int c = 0; c < 8; ++c) {
        int it = ibase + c; if (it >= Nit) it = Nit - 1; // clamp; masked at store
        x0[c] = r0[it]; x1[c] = r1[it];
      }
    }
#pragma unroll
    for (int c = 0; c < 8; ++c) lds_b[bnt][blb + c][bvg] = pack_bf16(x0[c], x1[c]);
  };

  int k0 = kb;
  for (; k0 + 32 <= ke; k0 += 32) {   // hot loop
    __syncthreads();
    const int k = k0 + 2 * kp;
    {
      f32x4 a0 = *(const f32x4*)(U + (size_t)k * 64 + adb);
      f32x4 a1 = *(const f32x4*)(U + (size_t)(k + 1) * 64 + adb);
      lds_a[amt][amb + 0 + alo][avg] = pack_bf16(a0.x, a1.x);
      lds_a[amt][amb + 1 + alo][avg] = pack_bf16(a0.y, a1.y);
      lds_a[amt][amb + 2 + alo][avg] = pack_bf16(a0.z, a1.z);
      lds_a[amt][amb + 3 + alo][avg] = pack_bf16(a0.w, a1.w);
    }
    stage_b(Adj + (size_t)k * Nit, Adj + (size_t)(k + 1) * Nit);
    __syncthreads();
    compute();
  }

  if (k0 < ke) {                      // peeled tail (K%32 != 0 splits only)
    __syncthreads();
    const int k = k0 + 2 * kp;
    {
      f32x4 a0 = {0.f,0.f,0.f,0.f}, a1 = {0.f,0.f,0.f,0.f};
      if (k < ke)     a0 = *(const f32x4*)(U + (size_t)k * 64 + adb);
      if (k + 1 < ke) a1 = *(const f32x4*)(U + (size_t)(k + 1) * 64 + adb);
      lds_a[amt][amb + 0 + alo][avg] = pack_bf16(a0.x, a1.x);
      lds_a[amt][amb + 1 + alo][avg] = pack_bf16(a0.y, a1.y);
      lds_a[amt][amb + 2 + alo][avg] = pack_bf16(a0.z, a1.z);
      lds_a[amt][amb + 3 + alo][avg] = pack_bf16(a0.w, a1.w);
    }
    {
      int ibase = n0 + bnb;
      float x0[8], x1[8];
      for (int c = 0; c < 8; ++c) {
        int it = ibase + c; if (it >= Nit) it = Nit - 1;
        x0[c] = (k < ke)     ? Adj[(size_t)k * Nit + it]       : 0.0f;
        x1[c] = (k + 1 < ke) ? Adj[(size_t)(k + 1) * Nit + it] : 0.0f;
      }
#pragma unroll
      for (int c = 0; c < 8; ++c) lds_b[bnt][blb + c][bvg] = pack_bf16(x0[c], x1[c]);
    }
    __syncthreads();
    compute();
  }

  float* Cp = Cpart + (size_t)blockIdx.y * ((size_t)Nit * 64);
  const int item = n0 + w * 16 + (lane & 15);
  if (item < Nit) {
#pragma unroll
    for (int mt = 0; mt < 4; ++mt) {
      int dbase = mt * 16 + ((lane & 16) ? 8 : 0);
#pragma unroll
      for (int r = 0; r < 8; ++r) Cp[(size_t)item * 64 + dbase + r] = acc[mt][r];
    }
  }
}

// ---------------------------------------------------------------------------
// Deterministic split-K reduction: next = sum_s part[s]; acc += next.
// ---------------------------------------------------------------------------
__global__ void gcn_reduce(const float* __restrict__ part, size_t stride, int S,
                           float* __restrict__ next, float* __restrict__ acc, int n) {
  int i = blockIdx.x * blockDim.x + threadIdx.x;
  if (i < n) {
    float s = 0.0f;
    for (int t = 0; t < S; ++t) s += part[(size_t)t * stride + i];
    next[i] = s;
    acc[i] += s;
  }
}

// ---------------------------------------------------------------------------
// Layer-0 feature projections (tiny; VALU is fine).
// ---------------------------------------------------------------------------
__global__ __launch_bounds__(64) void gcn_feat_user(
    const float* __restrict__ emb, const int* __restrict__ ridx,
    const int* __restrict__ tidx, const float* __restrict__ recW,
    const float* __restrict__ typW, const float* __restrict__ W,
    const float* __restrict__ bias, float* __restrict__ u0,
    float* __restrict__ out_u) {
  int m = blockIdx.x, d = threadIdx.x;
  __shared__ float f[80];
  f[d] = emb[(size_t)m * 64 + d];
  if (d < 8)       f[64 + d]       = recW[ridx[m] * 8 + d];
  else if (d < 16) f[72 + (d - 8)] = typW[tidx[m] * 8 + (d - 8)];
  __syncthreads();
  const float* wr = W + d * 80;
  float a = bias[d];
#pragma unroll 8
  for (int j = 0; j < 80; ++j) a += wr[j] * f[j];
  size_t idx = (size_t)m * 64 + d;
  u0[idx] = a; out_u[idx] = a;   // overwrite of d_out => deterministic replays
}

__global__ __launch_bounds__(64) void gcn_feat_item(
    const float* __restrict__ emb, const int* __restrict__ xidx,
    const float* __restrict__ resW, const float* __restrict__ W,
    const float* __restrict__ bias, float* __restrict__ i0,
    float* __restrict__ out_i) {
  int m = blockIdx.x, d = threadIdx.x;
  __shared__ float f[72];
  f[d] = emb[(size_t)m * 64 + d];
  if (d < 8) f[64 + d] = resW[xidx[m] * 8 + d];
  __syncthreads();
  const float* wr = W + d * 72;
  float a = bias[d];
#pragma unroll 8
  for (int j = 0; j < 72; ++j) a += wr[j] * f[j];
  size_t idx = (size_t)m * 64 + d;
  i0[idx] = a; out_i[idx] = a;
}

__global__ void gcn_scale(float* __restrict__ p, int n, float s) {
  int i = blockIdx.x * blockDim.x + threadIdx.x;
  if (i < n) p[i] *= s;
}

// ---------------------------------------------------------------------------
extern "C" void kernel_launch(void* const* d_in, const int* in_sizes, int n_in,
                              void* d_out, int out_size, void* d_ws, size_t ws_size,
                              hipStream_t stream) {
  const float* adj      = (const float*)d_in[0];
  const int*   ridx     = (const int*)  d_in[1];
  const int*   tidx     = (const int*)  d_in[2];
  const int*   xidx     = (const int*)  d_in[3];
  const float* user_emb = (const float*)d_in[4];
  const float* item_emb = (const float*)d_in[5];
  const float* recW     = (const float*)d_in[6];
  const float* typW     = (const float*)d_in[7];
  const float* resW     = (const float*)d_in[8];
  const float* Wu       = (const float*)d_in[9];
  const float* bu       = (const float*)d_in[10];
  const float* Wi       = (const float*)d_in[11];
  const float* bi       = (const float*)d_in[12];

  const int M = in_sizes[1];   // 20000 users
  const int I = in_sizes[3];   // 10000 items
  const size_t uN = (size_t)M * 64, iN = (size_t)I * 64;

  float* out_u = (float*)d_out;
  float* out_i = out_u + uN;

  float* u0 = (float*)d_ws;
  float* u1 = u0 + uN;
  float* i0 = u1 + uN;
  float* i1 = i0 + iN;
  float* upart = i1 + iN;

  // split-K factor from available workspace (deterministic per environment)
  int S = 1;
  {
    size_t words = ws_size / 4;
    size_t base  = 2 * uN + 2 * iN;
    if (words > base) {
      size_t s = (words - base) / (uN + iN);
      if (s < 1) s = 1; if (s > 8) s = 8;
      S = (int)s;
    }
  }
  float* ipart = upart + (size_t)S * uN;

  // k-span per split, in whole 32-wide K-steps
  const int steps_ab  = (I + 31) / 32;
  const int kspan_ab  = ((steps_ab + S - 1) / S) * 32;
  const int steps_atb = (M + 31) / 32;
  const int kspan_atb = ((steps_atb + S - 1) / S) * 32;

  gcn_feat_user<<<M, 64, 0, stream>>>(user_emb, ridx, tidx, recW, typW, Wu, bu, u0, out_u);
  gcn_feat_item<<<I, 64, 0, stream>>>(item_emb, xidx, resW, Wi, bi, i0, out_i);

  float* uc = u0; float* un = u1; float* ic = i0; float* ix = i1;
  const dim3 gu((M + 127) / 128, S);
  const dim3 gi((I + 127) / 128, S);
  for (int l = 0; l < 3; ++l) {
    gcn_gemm_ab <<<gu, 256, 0, stream>>>(adj, ic, upart, M, I, kspan_ab);
    gcn_gemm_atb<<<gi, 256, 0, stream>>>(adj, uc, ipart, M, I, kspan_atb);
    gcn_reduce<<<((int)uN + 255) / 256, 256, 0, stream>>>(upart, uN, S, un, out_u, (int)uN);
    gcn_reduce<<<((int)iN + 255) / 256, 256, 0, stream>>>(ipart, iN, S, ix, out_i, (int)iN);
    float* t;
    t = uc; uc = un; un = t;
    t = ic; ic = ix; ix = t;
  }

  const int total = (int)(uN + iN);
  gcn_scale<<<(total + 255) / 256, 256, 0, stream>>>((float*)d_out, total, 0.25f);
}
